// FramePrimerEncoder_56435870269524
// MI455X (gfx1250) — matrix operand order repaired and checked by reading, not compiled, not run on hardware
//
#include <hip/hip_runtime.h>
#include <hip/hip_bf16.h>

typedef unsigned short u16;
typedef unsigned int   u32;

typedef __attribute__((ext_vector_type(16))) __bf16 v16bf;
typedef __attribute__((ext_vector_type(8)))  float  v8f;

#define B_    2
#define C_    8
#define BINS_ 1024
#define W_    512
#define HEADS_ 4
#define DH_   256

// ---------------- helpers ----------------

__device__ __forceinline__ u16 f2bf(float f){
  u32 u = __float_as_uint(f);
  u32 r = u + 0x7fffu + ((u >> 16) & 1u);   // round-to-nearest-even
  return (u16)(r >> 16);
}

union ABFrag { float4 f[2]; v16bf v; };

__device__ __forceinline__ v8f wmma_bf16(const ABFrag& a, const ABFrag& b, v8f c){
  // D(16x16,f32) = A(16x32,bf16) x B(32x16,bf16) + C
  return __builtin_amdgcn_wmma_f32_16x16x32_bf16(false, a.v, false, b.v,
                                                 (short)0, c, false, false);
}

struct BMap { long long s1, s2; int div, mod; };
__device__ __forceinline__ long long boff(BMap m, int z){
  return (long long)((z / m.div) % m.mod) * m.s1 + (long long)(z % m.div) * m.s2;
}

// ---------------- batched GEMM via WMMA ----------------
// C[M,N] = A[M,K](bf16, row-major lda) * Bt[N,K]^T (bf16, row-major ldb), f32 accum.
// Block: 256 thr = 8 waves (2 x 4); wave tile 64x32 (4x2 WMMA frags, 8 accums);
// block tile 128x128. 8 WMMAs per 12 b128 fragment loads per K-step.
// EPI: 0 = f32 store, 1 = f32*scale, 2 = bf16 store, 3 = f32 store + residual,
//      4 = relu^2 -> bf16, stored TRANSPOSED (addr = col*ldc + row), packed b128.

template<int EPI>
__device__ __forceinline__ void store_frag(void* Cv, const float* __restrict__ R,
                                           long long cbase, int ldc, int mm, int nn,
                                           int lr, int half, v8f cc, float scale)
{
  if (EPI == 4){
    union { u16 h[8]; uint4 q; } pk;
    #pragma unroll
    for (int r = 0; r < 8; ++r){
      float f = cc[r];
      f = f > 0.f ? f * f : 0.f;
      pk.h[r] = f2bf(f);
    }
    *(uint4*)((u16*)Cv + cbase + (size_t)(nn + lr) * ldc + mm + 8 * half) = pk.q;
  } else {
    #pragma unroll
    for (int r = 0; r < 8; ++r){
      long long off = cbase + (long long)(mm + r + 8 * half) * ldc + nn + lr;
      float f = cc[r];
      if (EPI == 0) ((float*)Cv)[off] = f;
      if (EPI == 1) ((float*)Cv)[off] = f * scale;
      if (EPI == 2) ((u16*)Cv)[off]   = f2bf(f);
      if (EPI == 3) ((float*)Cv)[off] = f + R[off];
    }
  }
}

template<int EPI>
__global__ __launch_bounds__(256) void gemm_bf16_k(
    const u16* __restrict__ A, const u16* __restrict__ Bt, void* __restrict__ Cv,
    const float* __restrict__ R, int K, int lda, int ldb, int ldc,
    BMap ma, BMap mb, BMap mc, float scale)
{
  const int z = blockIdx.z;
  A  += boff(ma, z);
  Bt += boff(mb, z);
  const long long cbase = boff(mc, z);

  const int wid  = threadIdx.x >> 5;
  const int lane = threadIdx.x & 31;
  const int half = lane >> 4;
  const int lr   = lane & 15;
  const int m0 = blockIdx.x * 128 + (wid & 1)  * 64;   // wave tile: 64 rows
  const int n0 = blockIdx.y * 128 + (wid >> 1) * 32;   //            32 cols

  const u16* ap[4];
  const u16* bp[2];
  #pragma unroll
  for (int i = 0; i < 4; ++i) ap[i] = A  + (size_t)(m0 + 16*i + lr) * lda;
  #pragma unroll
  for (int j = 0; j < 2; ++j) bp[j] = Bt + (size_t)(n0 + 16*j + lr) * ldb;

  v8f acc[4][2];
  #pragma unroll
  for (int i = 0; i < 4; ++i)
    #pragma unroll
    for (int j = 0; j < 2; ++j) acc[i][j] = v8f{};

  for (int k = 0; k < K; k += 32){
    ABFrag af[4], bf[2];
    // A frag (16x32 bf16): lanes0-15 K=8h..8h+7 / 16+8h.., per ISA 7.12.2
    #pragma unroll
    for (int i = 0; i < 4; ++i){
      af[i].f[0] = *(const float4*)(ap[i] + k + 8*half);
      af[i].f[1] = *(const float4*)(ap[i] + k + 16 + 8*half);
    }
    // B frag (32x16 bf16): lane half h holds K 8h..8h+7 / 16+8h.., matching A
    #pragma unroll
    for (int j = 0; j < 2; ++j){
      bf[j].f[0] = *(const float4*)(bp[j] + k + 8*half);
      bf[j].f[1] = *(const float4*)(bp[j] + k + 16 + 8*half);
    }
    __builtin_prefetch(ap[0] + k + 128, 0, 0);
    __builtin_prefetch(bp[0] + k + 128, 0, 0);
    #pragma unroll
    for (int i = 0; i < 4; ++i)
      #pragma unroll
      for (int j = 0; j < 2; ++j)
        acc[i][j] = wmma_bf16(af[i], bf[j], acc[i][j]);
  }

  #pragma unroll
  for (int i = 0; i < 4; ++i)
    #pragma unroll
    for (int j = 0; j < 2; ++j)
      store_frag<EPI>(Cv, R, cbase, ldc, m0 + 16*i, n0 + 16*j, lr, half,
                      acc[i][j], scale);
}

// ---------------- f32 -> bf16 weight convert ----------------
__global__ __launch_bounds__(256) void cvt_bf16_k(const float* __restrict__ s,
                                                  u16* __restrict__ d, int n)
{
  int i = blockIdx.x * 256 + threadIdx.x;
  if (i < n) d[i] = f2bf(s[i]);
}

// ---------------- frame norm (LN over BINS axis), writes z^T bf16 ----------------
// zT layout: (b,c,w,h), so GEMM B-fragments read contiguous K=h.
__global__ __launch_bounds__(256) void frame_norm_k(const float* __restrict__ x,
    const float* __restrict__ w, const float* __restrict__ b, u16* __restrict__ zT)
{
  int col  = blockIdx.x * 256 + threadIdx.x;   // (b*C+c)*512 + wcol ; 8192 total
  int bc   = col >> 9;
  int wcol = col & 511;
  const float* xp = x + (size_t)bc * (BINS_*W_) + wcol;
  float s = 0.f, s2 = 0.f;
  for (int h = 0; h < BINS_; ++h){
    float v = xp[(size_t)h * W_];
    s += v; s2 += v * v;
  }
  float mu  = s * (1.f / BINS_);
  float var = s2 * (1.f / BINS_) - mu * mu;
  float inv = rsqrtf(var + 1e-5f);
  u32* zp = (u32*)(zT + (size_t)col * BINS_);
  for (int h = 0; h < BINS_; h += 2){
    float v0 = (xp[(size_t)h     * W_] - mu) * inv * w[h]   + b[h];
    float v1 = (xp[(size_t)(h+1) * W_] - mu) * inv * w[h+1] + b[h+1];
    zp[h >> 1] = (u32)f2bf(v0) | ((u32)f2bf(v1) << 16);
  }
}

// ---------------- 1x3 channel-mixing conv; writes f32 and bf16 copies ----------------
__global__ __launch_bounds__(256) void conv1x3_k(const float* __restrict__ p,
    const float* __restrict__ wgt, const float* __restrict__ bias,
    float* __restrict__ pc, u16* __restrict__ pcb)
{
  size_t idx = (size_t)blockIdx.x * 256 + threadIdx.x; // ((b*8+o)*1024+h)*512+w
  int w = (int)(idx & 511);
  size_t t = idx >> 9;
  int h = (int)(t & 1023); t >>= 10;
  int o = (int)(t & 7);
  int b = (int)(t >> 3);
  float acc = bias[o];
  const float* xb = p + ((size_t)b * C_) * (BINS_*W_) + (size_t)h * W_;
  #pragma unroll
  for (int i = 0; i < C_; ++i){
    const float* xi = xb + (size_t)i * (BINS_*W_);
    float x0 = (w > 0)   ? xi[w-1] : 0.f;
    float x1 = xi[w];
    float x2 = (w < 511) ? xi[w+1] : 0.f;
    const float* wk = wgt + (o * C_ + i) * 3;
    acc += x0 * wk[0] + x1 * wk[1] + x2 * wk[2];
  }
  pc[idx]  = acc;
  pcb[idx] = f2bf(acc);
}

// ---------------- rotary -> q/k bf16, layout (b,c,n,w,d) ----------------
__global__ __launch_bounds__(256) void rotary_k(const float* __restrict__ pc,
                                                u16* __restrict__ qb)
{
  size_t idx = (size_t)blockIdx.x * 256 + threadIdx.x; // ((bcn*128 + j)*512 + w)
  int w   = (int)(idx & 511);
  int j   = (int)((idx >> 9) & 127);
  int bcn = (int)(idx >> 16);
  int n   = bcn & 3;
  int bc  = bcn >> 2;
  int d0  = 2 * j;
  const float* src = pc + ((size_t)bc * BINS_ + n * DH_ + d0) * W_ + w;
  float t0 = src[0], t1 = src[W_];
  float o0, o1;
  if (d0 < 128){                                   // rotated half (rot = dh/2)
    float ang = (float)w * __powf(10000.f, -(float)d0 * (1.f / 128.f));
    float sn, cs; __sincosf(ang, &sn, &cs);
    o0 = t0 * cs - t1 * sn;
    o1 = t1 * cs + t0 * sn;
  } else { o0 = t0; o1 = t1; }                     // pass-through half
  ((u32*)qb)[((size_t)bcn * 512 + w) * 128 + j] =
      (u32)f2bf(o0) | ((u32)f2bf(o1) << 16);
}

// ---------------- row softmax (512 cols), f32 in -> bf16 out ----------------
__global__ __launch_bounds__(256) void softmax_k(const float* __restrict__ s,
                                                 u16* __restrict__ p)
{
  int wid = threadIdx.x >> 5, lane = threadIdx.x & 31;
  size_t row = (size_t)blockIdx.x * 8 + wid;       // 32768 rows total
  const float* sr = s + row * 512;
  float v[16], mx = -3.0e38f;
  #pragma unroll
  for (int i = 0; i < 16; ++i){ v[i] = sr[lane + i*32]; mx = fmaxf(mx, v[i]); }
  #pragma unroll
  for (int off = 16; off >= 1; off >>= 1) mx = fmaxf(mx, __shfl_xor(mx, off, 32));
  float sum = 0.f;
  #pragma unroll
  for (int i = 0; i < 16; ++i){ v[i] = __expf(v[i] - mx); sum += v[i]; }
  #pragma unroll
  for (int off = 16; off >= 1; off >>= 1) sum += __shfl_xor(sum, off, 32);
  float inv = 1.f / sum;
  u16* pr = p + row * 512;
  #pragma unroll
  for (int i = 0; i < 16; ++i) pr[lane + i*32] = f2bf(v[i] * inv);
}

// ---------------- orchestration ----------------

extern "C" void kernel_launch(void* const* d_in, const int* in_sizes, int n_in,
                              void* d_out, int out_size, void* d_ws, size_t ws_size,
                              hipStream_t stream)
{
  (void)in_sizes; (void)n_in; (void)out_size; (void)ws_size;

  const float* x      = (const float*)d_in[0];
  const float* ln1_w  = (const float*)d_in[1];
  const float* ln1_b  = (const float*)d_in[2];
  const float* qmclw  = (const float*)d_in[3];
  const float* qconvw = (const float*)d_in[4];
  const float* qconvb = (const float*)d_in[5];
  const float* omclw  = (const float*)d_in[6];
  const float* ln2_w  = (const float*)d_in[7];
  const float* ln2_b  = (const float*)d_in[8];
  const float* lin1w  = (const float*)d_in[9];
  const float* lin2w  = (const float*)d_in[10];
  float* out = (float*)d_out;

  char* ws = (char*)d_ws;
  const size_t MB = 1ull << 20;
  u16*   wq_bf  = (u16*)  (ws +   0*MB);   // 16MB  (C,1024,1024) bf16
  u16*   wo_bf  = (u16*)  (ws +  16*MB);   // 16MB
  u16*   l1_bf  = (u16*)  (ws +  32*MB);   // 64MB  (C,4096,1024)
  u16*   l2_bf  = (u16*)  (ws +  96*MB);   // 64MB  (C,1024,4096)
  u16*   zT     = (u16*)  (ws + 160*MB);   // 16MB  (b,c,w,h) bf16  (reused for LN2)
  float* p      = (float*)(ws + 176*MB);   // 32MB  mcl1 out (b,c,h,w)     } lifetimes
  float* scores = (float*)(ws + 176*MB);   // 64MB  (b,c,n,q,k)            } overlap
  u16*   gT     = (u16*)  (ws + 176*MB);   // 64MB  (b,c,w,4096) bf16      } safely
  float* pc     = (float*)(ws + 208*MB);   // 32MB  conv out (b,c,h,w)  (inside scores region; dead first)
  u16*   pcb    = (u16*)  (ws + 240*MB);   // 16MB  conv out bf16 == V^T per head
  u16*   qb     = (u16*)  (ws + 256*MB);   // 16MB  rotary(q)=rotary(k), (b,c,n,w,d) bf16
  u16*   Pbf    = (u16*)  (ws + 272*MB);   // 32MB  softmax probs bf16
  u16*   aT     = (u16*)  (ws + 304*MB);   // 16MB  attn out, (b,c,w,h) bf16
  float* x1     = (float*)(ws + 320*MB);   // 32MB  x + attn   -> total 352MB

  const int BIG = 1 << 30;

  // weight conversions f32 -> bf16
  cvt_bf16_k<<<(8388608 + 255)/256, 256, 0, stream>>>(qmclw, wq_bf, 8388608);
  cvt_bf16_k<<<(8388608 + 255)/256, 256, 0, stream>>>(omclw, wo_bf, 8388608);
  cvt_bf16_k<<<(33554432 + 255)/256, 256, 0, stream>>>(lin1w, l1_bf, 33554432);
  cvt_bf16_k<<<(33554432 + 255)/256, 256, 0, stream>>>(lin2w, l2_bf, 33554432);

  // z = LN1(x), transposed bf16
  frame_norm_k<<<32, 256, 0, stream>>>(x, ln1_w, ln1_b, zT);

  { // GEMM1: p = Wq[c] @ z   (M=1024,N=512,K=1024, batch b*c=16)
    BMap ma{1024LL*1024, 0, 1, 8}, mb{512LL*1024, 0, 1, BIG}, mc{1024LL*512, 0, 1, BIG};
    gemm_bf16_k<0><<<dim3(8, 4, 16), 256, 0, stream>>>(wq_bf, zT, p, nullptr,
        1024, 1024, 1024, 512, ma, mb, mc, 1.f);
  }
  // conv over W + channel mix; writes V (f32 for rotary, bf16 for PV GEMM)
  conv1x3_k<<<8388608/256, 256, 0, stream>>>(p, qconvw, qconvb, pc, pcb);
  // rotary (q == k)
  rotary_k<<<(64*128*512)/256, 256, 0, stream>>>(pc, qb);

  { // GEMM2: scores = q @ q^T / 32  (M=512,N=512,K=256, batch b*c*heads=64)
    BMap ma{512LL*256, 0, 1, BIG}, mb{512LL*256, 0, 1, BIG}, mc{512LL*512, 0, 1, BIG};
    gemm_bf16_k<1><<<dim3(4, 4, 64), 256, 0, stream>>>(qb, qb, scores, nullptr,
        256, 256, 256, 512, ma, mb, mc, 1.f/32.f);
  }
  softmax_k<<<(64*512)/8, 256, 0, stream>>>(scores, Pbf);

  { // GEMM3: a = P @ V  (M=512,N=256,K=512, batch 64); store bf16 into (b,c,w,h)
    BMap ma{512LL*512, 0, 1, BIG}, mb{256LL*512, 0, 1, BIG},
         mc{512LL*1024, 256, 4, BIG};                 // off = (z/4)*524288 + head*256
    gemm_bf16_k<2><<<dim3(4, 2, 64), 256, 0, stream>>>(Pbf, pcb, aT, nullptr,
        512, 512, 512, 1024, ma, mb, mc, 1.f);
  }
  { // GEMM4: x1 = x + Wo[c] @ a  (M=1024,N=512,K=1024, batch 16)
    BMap ma{1024LL*1024, 0, 1, 8}, mb{512LL*1024, 0, 1, BIG}, mc{1024LL*512, 0, 1, BIG};
    gemm_bf16_k<3><<<dim3(8, 4, 16), 256, 0, stream>>>(wo_bf, aT, x1, x,
        1024, 1024, 1024, 512, ma, mb, mc, 1.f);
  }

  // z2 = LN2(x1), transposed bf16 (reuse zT)
  frame_norm_k<<<32, 256, 0, stream>>>(x1, ln2_w, ln2_b, zT);

  { // GEMM5: g = relu(L1[c] @ z2)^2, stored transposed bf16 (M=4096,N=512,K=1024)
    BMap ma{4096LL*1024, 0, 1, 8}, mb{512LL*1024, 0, 1, BIG}, mc{512LL*4096, 0, 1, BIG};
    gemm_bf16_k<4><<<dim3(32, 4, 16), 256, 0, stream>>>(l1_bf, zT, gT, nullptr,
        1024, 1024, 1024, 4096, ma, mb, mc, 1.f);
  }
  { // GEMM6: out = x1 + L2[c] @ g  (M=1024,N=512,K=4096)
    BMap ma{1024LL*4096, 0, 1, 8}, mb{512LL*4096, 0, 1, BIG}, mc{1024LL*512, 0, 1, BIG};
    gemm_bf16_k<3><<<dim3(8, 4, 16), 256, 0, stream>>>(l2_bf, gT, out, x1,
        4096, 4096, 4096, 512, ma, mb, mc, 1.f);
  }
}
